// GCN_65386582114681
// MI455X (gfx1250) — matrix operand order repaired and checked by reading, not compiled, not run on hardware
//
#include <hip/hip_runtime.h>
#include <hip/hip_bf16.h>
#include <stdint.h>

// ---------------------------------------------------------------------------
// GCN forward (2 layers) for gfx1250.
//  - Activations pre-packed to bf16 pairs -> GEMM A fragments are pure
//    global_load_b128; W staged transposed in LDS -> B fragments are pure
//    ds_load_b128.  4x v_wmma_f32_16x16x32_bf16 per 16x16 output tile.
//  - Fused epilogue writes H and the bias+self-loop-initialized target.
//  - Edge scatter: one wave per edge, hardware global_atomic_add_f32.
// Assumes n_rows % 16 == 0 (N = 100000 here).
// ---------------------------------------------------------------------------

typedef __bf16 bf16_t;
typedef __attribute__((ext_vector_type(16))) bf16_t   v16bf;
typedef __attribute__((ext_vector_type(8)))  float    v8f;
typedef __attribute__((ext_vector_type(8)))  unsigned v8u;

static __device__ __forceinline__ unsigned pack2_bf16(float a, float b) {
    unsigned short ua = __builtin_bit_cast(unsigned short, (bf16_t)a);
    unsigned short ub = __builtin_bit_cast(unsigned short, (bf16_t)b);
    return (unsigned)ua | ((unsigned)ub << 16);
}

// ---- degree / symmetric normalization -------------------------------------
__global__ void k_init_deg(int* __restrict__ deg, int n) {
    int i = blockIdx.x * blockDim.x + threadIdx.x;
    if (i < n) deg[i] = 1;                      // self-loop contributes 1
}

__global__ void k_count_deg(const int* __restrict__ cols, int* __restrict__ deg, int e) {
    int i = blockIdx.x * blockDim.x + threadIdx.x;
    if (i < e) atomicAdd(&deg[cols[i]], 1);
}

__global__ void k_make_dis(float* __restrict__ disf, int n) {
    int i = blockIdx.x * blockDim.x + threadIdx.x;
    if (i < n) {
        int d = ((const int*)disf)[i];          // deg >= 1 always
        disf[i] = rsqrtf((float)d);
    }
}

// ---- fp32 -> packed bf16 conversion kernels -------------------------------
__global__ void k_cvt_bf16(const float* __restrict__ in, unsigned* __restrict__ out,
                           int total2) {
    int i = blockIdx.x * blockDim.x + threadIdx.x;
    if (i >= total2) return;
    float2 v = ((const float2*)in)[i];
    out[i] = pack2_bf16(v.x, v.y);
}

__global__ void k_relu_cvt_bf16(const float* __restrict__ in, unsigned* __restrict__ out,
                                int total2) {
    int i = blockIdx.x * blockDim.x + threadIdx.x;
    if (i >= total2) return;
    float2 v = ((const float2*)in)[i];
    out[i] = pack2_bf16(fmaxf(v.x, 0.0f), fmaxf(v.y, 0.0f));
}

// ---- fused GEMM + epilogue ------------------------------------------------
// H[n_rows x FOUT]    = A[n_rows x 128 bf16] @ W[128 x FOUT f32]
// Init[n_rows x FOUT] = bias[f] + H * dis[v]^2      (bias + self-loop term)
// Block = 8 waves x 16 rows.  Abf: packed bf16 pairs, row stride 64 uints.
template <int FOUT>
__global__ __launch_bounds__(256)
void k_gemm_fused(const unsigned* __restrict__ Abf, const float* __restrict__ W,
                  const float* __restrict__ dis, const float* __restrict__ bias,
                  float* __restrict__ H, float* __restrict__ Init, int n_rows) {
    // Transposed, padded: row pitch 68 uints keeps 16B alignment (68%4==0)
    // and staggers banks (4*col mod 64).
    __shared__ unsigned WldsT[FOUT][68];        // [col][K/2] packed bf16x2

    const int tid = threadIdx.x;

    // stage all of W: 64*FOUT packed uints, 256 threads
    constexpr int TOTAL = 64 * FOUT;
    #pragma unroll
    for (int t = 0; t < TOTAL / 256; ++t) {
        int idx = tid + t * 256;
        int k2  = idx / FOUT;
        int n   = idx % FOUT;
        float w0 = W[(size_t)(2 * k2    ) * FOUT + n];
        float w1 = W[(size_t)(2 * k2 + 1) * FOUT + n];
        WldsT[n][k2] = pack2_bf16(w0, w1);
    }
    __syncthreads();

    const int wave  = tid >> 5;
    const int lane  = tid & 31;
    const int group = lane >> 4;                // 0: lanes 0-15, 1: lanes 16-31
    const int m     = lane & 15;
    const int row0  = blockIdx.x * 128 + wave * 16;
    if (row0 >= n_rows) return;                 // wave-uniform; n_rows % 16 == 0

    const unsigned* arow = Abf + (size_t)(row0 + m) * 64;
    __builtin_prefetch(arow, 0, 0);             // global_prefetch_b8

    // A fragments (16-bit A 16x32 layout): lane m / group g holds K pairs
    // {0..7,16..23} + 8*g  ->  uints [kt*16+g*4 .. +3] and [kt*16+8+g*4 .. +3]
    v8u araw[4];
    #pragma unroll
    for (int kt = 0; kt < 4; ++kt) {
        uint4 q0 = *(const uint4*)(arow + kt * 16 + group * 4);
        uint4 q1 = *(const uint4*)(arow + kt * 16 + 8 + group * 4);
        araw[kt][0] = q0.x; araw[kt][1] = q0.y;
        araw[kt][2] = q0.z; araw[kt][3] = q0.w;
        araw[kt][4] = q1.x; araw[kt][5] = q1.y;
        araw[kt][6] = q1.z; araw[kt][7] = q1.w;
    }

    // self-loop scale per output row (uniform across the 16 lanes of a group)
    float s2arr[8];
    #pragma unroll
    for (int i = 0; i < 8; ++i) {
        float s = dis[row0 + i + 8 * group];
        s2arr[i] = s * s;
    }

    #pragma unroll
    for (int ct = 0; ct < FOUT / 16; ++ct) {
        const int col0 = ct * 16;
        const unsigned* wrow = &WldsT[col0 + m][0];
        v8f acc = {};
        #pragma unroll
        for (int kt = 0; kt < 4; ++kt) {
            // B (32x16 bf16): lane group g covers K = {0..7,16..23} + 8g -> 8 uints
            uint4 p0 = *(const uint4*)(wrow + kt * 16 + group * 4);
            uint4 p1 = *(const uint4*)(wrow + kt * 16 + 8 + group * 4);
            v8u braw;
            braw[0] = p0.x; braw[1] = p0.y; braw[2] = p0.z; braw[3] = p0.w;
            braw[4] = p1.x; braw[5] = p1.y; braw[6] = p1.z; braw[7] = p1.w;
            v16bf a = __builtin_bit_cast(v16bf, araw[kt]);
            v16bf b = __builtin_bit_cast(v16bf, braw);
            acc = __builtin_amdgcn_wmma_f32_16x16x32_bf16(
                      false, a, false, b, (short)0, acc, false, false);
        }
        // C/D layout: lane group g, VGPR i -> M = row0 + i + 8g, N = m
        const float bv = bias[col0 + m];
        #pragma unroll
        for (int i = 0; i < 8; ++i) {
            size_t off = (size_t)(row0 + i + 8 * group) * FOUT + col0 + m;
            float  v   = acc[i];
            H[off]    = v;
            Init[off] = bv + v * s2arr[i];
        }
    }
}

// ---- edge scatter: one wave per edge, hardware fp32 atomics ---------------
template <int F>
__global__ __launch_bounds__(256)
void k_aggregate(const int* __restrict__ rows, const int* __restrict__ cols,
                 const float* __restrict__ dis, const float* __restrict__ H,
                 float* __restrict__ Out, int e) {
    int gid  = blockIdx.x * blockDim.x + threadIdx.x;
    int edge = gid >> 5;
    int lane = gid & 31;
    if (edge >= e) return;
    int r = rows[edge];
    int c = cols[edge];
    float coef = dis[r] * dis[c];
    constexpr int V = F / 32;                   // 4 (F=128) or 2 (F=64)
    const float* src = H + (size_t)r * F + lane * V;
    float*       dst = Out + (size_t)c * F + lane * V;
    if constexpr (V == 4) {
        float4 v = *(const float4*)src;
        unsafeAtomicAdd(dst + 0, v.x * coef);
        unsafeAtomicAdd(dst + 1, v.y * coef);
        unsafeAtomicAdd(dst + 2, v.z * coef);
        unsafeAtomicAdd(dst + 3, v.w * coef);
    } else {
        float2 v = *(const float2*)src;
        unsafeAtomicAdd(dst + 0, v.x * coef);
        unsafeAtomicAdd(dst + 1, v.y * coef);
    }
}

// ---------------------------------------------------------------------------
extern "C" void kernel_launch(void* const* d_in, const int* in_sizes, int n_in,
                              void* d_out, int out_size, void* d_ws, size_t ws_size,
                              hipStream_t stream) {
    const float* x  = (const float*)d_in[0];   // N x 128
    const int*   ei = (const int*)  d_in[1];   // 2 x E
    const float* W1 = (const float*)d_in[2];   // 128 x 128
    const float* b1 = (const float*)d_in[3];   // 128
    const float* W2 = (const float*)d_in[4];   // 128 x 64
    const float* b2 = (const float*)d_in[5];   // 64
    float* out = (float*)d_out;                // N x 64

    const int N = in_sizes[0] / 128;
    const int E = in_sizes[1] / 2;
    const int* rows = ei;                      // sources
    const int* cols = ei + E;                  // targets

    float*    ws   = (float*)d_ws;
    float*    dis  = ws;                          // N floats (int degrees first)
    float*    bufA = ws + N;                      // N x 128 f32: H (gather src)
    float*    bufB = bufA + (size_t)N * 128;      // N x 128 f32: aggregation dst
    unsigned* Xbf  = (unsigned*)(bufB + (size_t)N * 128);  // N x 64 packed bf16
    unsigned* Hbf  = Xbf + (size_t)N * 64;                 // N x 64 packed bf16

    const int T = 256;

    // normalization coefficients
    k_init_deg <<<(N + T - 1) / T, T, 0, stream>>>((int*)dis, N);
    k_count_deg<<<(E + T - 1) / T, T, 0, stream>>>(cols, (int*)dis, E);
    k_make_dis <<<(N + T - 1) / T, T, 0, stream>>>(dis, N);

    const int aggBlocks  = (int)(((long long)E * 32 + T - 1) / T);
    const int gemmBlocks = (N + 127) / 128;

    // layer 1: Xbf = bf16(X); bufA = X@W1; bufB = b1 + self; scatter; relu->bf16
    k_cvt_bf16<<<(N * 64 + T - 1) / T, T, 0, stream>>>(x, Xbf, N * 64);
    k_gemm_fused<128><<<gemmBlocks, T, 0, stream>>>(Xbf, W1, dis, b1, bufA, bufB, N);
    k_aggregate<128><<<aggBlocks, T, 0, stream>>>(rows, cols, dis, bufA, bufB, E);
    k_relu_cvt_bf16<<<(N * 64 + T - 1) / T, T, 0, stream>>>(bufB, Hbf, N * 64);

    // layer 2: bufA = relu(H1)@W2; out = b2 + self; scatter
    k_gemm_fused<64><<<gemmBlocks, T, 0, stream>>>(Hbf, W2, dis, b2, bufA, out, N);
    k_aggregate<64><<<aggBlocks, T, 0, stream>>>(rows, cols, dis, bufA, out, E);
}
